// TexualEmbeddingLayer1_56831007261325
// MI455X (gfx1250) — matrix operand order repaired and checked by reading, not compiled, not run on hardware
//
#include <hip/hip_runtime.h>
#include <hip/hip_bf16.h>

typedef __attribute__((ext_vector_type(2)))  float    v2f;
typedef __attribute__((ext_vector_type(8)))  float    v8f;
typedef __attribute__((ext_vector_type(8)))  _Float16 v8h;
typedef __attribute__((ext_vector_type(16))) _Float16 v16h;
typedef __attribute__((ext_vector_type(4)))  unsigned u32x4;
typedef __attribute__((ext_vector_type(8)))  unsigned u32x8;

#define NB    2048
#define S_    77
#define DIN   512
#define DEMB  1024
#define SEL   10
#define MPAD  16
#define LDSF  516   // f32 LDS row stride: 512 data + 2 DW pad per 256 DW (TDM pad scheme)
#define LDSH  520   // f16 LDS row stride (elements): 1040B rows, 16B aligned

#if defined(__has_builtin)
# if __has_builtin(__builtin_amdgcn_wmma_f32_16x16x4_f32)
#  define HAVE_WMMA_F32X4 1
# endif
#endif

__device__ __forceinline__ unsigned lds_off(const void* p) {
  // generic pointer into the LDS aperture keeps the allocation-relative
  // byte offset in the low 32 bits (CDNA5 aperture mapping)
  return (unsigned)(size_t)p;
}

// ---------------------------------------------------------------- prep ------
__global__ void k_prep(const float* __restrict__ Wlin, _Float16* __restrict__ Wlin16,
                       float* __restrict__ gsum, float* __restrict__ gsq) {
  int i = blockIdx.x * 256 + threadIdx.x;
  if (i < DEMB * DIN) Wlin16[i] = (_Float16)Wlin[i];
  if (i < DIN) { gsum[i] = 0.0f; gsq[i] = 0.0f; }
}

// ------------------------------------------- topk + gather + normalize ------
__global__ __launch_bounds__(128)
void k_topk_gather(const float* __restrict__ emb, const int* __restrict__ ids,
                   const float* __restrict__ attn, float* __restrict__ feats,
                   int* __restrict__ lv) {
  const int b = blockIdx.x, tid = threadIdx.x;
  __shared__ int   sFirst, sNnz;
  __shared__ float sval[76];
  __shared__ float rv[128];
  __shared__ int   ri[128];
  __shared__ int   ssel[SEL];
  __shared__ float sred[128];

  if (tid == 0) { sFirst = S_; sNnz = 0; }
  __syncthreads();
  if (tid < S_) {
    int v = ids[b * S_ + tid];
    if (v == 0) atomicMin(&sFirst, tid);
    else        atomicAdd(&sNnz, 1);
  }
  __syncthreads();
  const int L = min(sFirst, 76);

  if (tid < 76) {
    float v = attn[((size_t)b * S_ + L) * S_ + 1 + tid];
    sval[tid] = (tid < L - 1) ? v : -INFINITY;
  }
  __syncthreads();

  // 10x argmax with lax.top_k tie rule: larger value wins; tie -> smaller index
  for (int it = 0; it < SEL; ++it) {
    rv[tid] = (tid < 76) ? sval[tid] : -INFINITY;
    ri[tid] = (tid < 76) ? tid : 75;
    __syncthreads();
    for (int off = 64; off > 0; off >>= 1) {
      if (tid < off) {
        float v2 = rv[tid + off]; int i2 = ri[tid + off];
        if (v2 > rv[tid] || (v2 == rv[tid] && i2 < ri[tid])) { rv[tid] = v2; ri[tid] = i2; }
      }
      __syncthreads();
    }
    if (tid == 0) { ssel[it] = ri[0]; sval[ri[0]] = -INFINITY; }
    __syncthreads();
  }

  // gather + L2 normalize; pad rows [10,16) with zeros for 16-row WMMA tiles
  for (int i = 0; i < SEL; ++i) {
    const int s = ssel[i] + 1;
    const float4 x = ((const float4*)(emb + ((size_t)b * S_ + s) * DIN))[tid];
    sred[tid] = x.x * x.x + x.y * x.y + x.z * x.z + x.w * x.w;
    __syncthreads();
    for (int off = 64; off > 0; off >>= 1) {
      if (tid < off) sred[tid] += sred[tid + off];
      __syncthreads();
    }
    const float inv = 1.0f / (sqrtf(sred[0]) + 1e-8f);
    __syncthreads();
    float4 y; y.x = x.x * inv; y.y = x.y * inv; y.z = x.z * inv; y.w = x.w * inv;
    ((float4*)(feats + ((size_t)b * MPAD + i) * DIN))[tid] = y;
  }
  const float4 z = {0.f, 0.f, 0.f, 0.f};
  for (int i = SEL; i < MPAD; ++i)
    ((float4*)(feats + ((size_t)b * MPAD + i) * DIN))[tid] = z;
  if (tid == 0) lv[b] = min(sNnz - 2, 30);
}

// --------------------------------------------- GEMM1: h = feats @ W0^T ------
// A tile staged by the Tensor Data Mover: 2D descriptor, tile 512x16 f32,
// LDS padding 2 DW per 256 DW -> row stride 516 (conflict-free column reads).
__global__ __launch_bounds__(256)
void k_gemm1(const float* __restrict__ feats, const float* __restrict__ W0,
             const float* __restrict__ b0, float* __restrict__ h) {
  const int b = blockIdx.x;
  __shared__ float lA[MPAD * LDSF];

  if (threadIdx.x < 32) {                       // wave 0 issues the TDM op
    const unsigned long long ga =
        (unsigned long long)(size_t)(feats + (size_t)b * MPAD * DIN);
    u32x4 g0;
    g0[0] = 1u;                                 // count=1, user descriptor
    g0[1] = lds_off(&lA[0]);                    // lds_addr (bytes)
    g0[2] = (unsigned)(ga & 0xffffffffu);       // global_addr[31:0]
    g0[3] = (unsigned)((ga >> 32) & 0x01ffffffu) | (2u << 30);  // addr[56:32] | type=2
    u32x8 g1;
    g1[0] = (2u << 16)          // data_size = 4B
          | (1u << 20)          // pad_enable
          | (7u << 22)          // pad_interval: 256 DWORDs
          | (1u << 25);         // pad_amount: 2 DWORDs
    g1[1] = (DIN & 0xffffu) << 16;              // tensor_dim0[15:0]=512
    g1[2] = (MPAD & 0xffffu) << 16;             // dim0[31:16]=0 | tensor_dim1[15:0]=16
    g1[3] = (DIN & 0xffffu) << 16;              // dim1[31:16]=0 | tile_dim0=512
    g1[4] = MPAD;                               // tile_dim1=16 | tile_dim2=0
    g1[5] = DIN;                                // tensor_dim0_stride[31:0]=512
    g1[6] = 0u;                                 // stride0[47:32]=0 | stride1[15:0]=0
    g1[7] = 0u;
    asm volatile("tensor_load_to_lds %0, %1" :: "s"(g0), "s"(g1) : "memory");
    __builtin_amdgcn_s_wait_tensorcnt(0);
  }
  __syncthreads();

  const int wave = threadIdx.x >> 5, lane = threadIdx.x & 31;
  const int nt = blockIdx.y * 8 + wave;           // 32 n-tiles total
  const int n  = nt * 16 + (lane & 15);
  const int m  = lane & 15;
  const int k0 = (lane < 16) ? 0 : 2;
  const float* wrow = W0 + (size_t)n * DIN;

  v8f c = {};
  for (int kb = 0; kb < DIN; kb += 4) {
    if ((kb & 63) == 0) __builtin_prefetch(wrow + kb + 64, 0, 3);
    const int pad = (kb >> 8) << 1;             // TDM pad: +2 DW past K=256
#ifdef HAVE_WMMA_F32X4
    v2f a  = *(const v2f*)(&lA[m * LDSF + kb + k0 + pad]);
    v2f bb = *(const v2f*)(wrow + kb + k0);
    c = __builtin_amdgcn_wmma_f32_16x16x4_f32(false, a, false, bb, (short)0, c,
                                              false, false);
#else
    const int roff = (lane < 16) ? 0 : 8;
    for (int kk = 0; kk < 4; ++kk) {
      const float bv = wrow[kb + kk];
      for (int j = 0; j < 8; ++j) c[j] += lA[(j + roff) * LDSF + kb + kk + pad] * bv;
    }
#endif
  }
  const float bias = b0[n];
  const int roff = (lane < 16) ? 0 : 8;
  for (int j = 0; j < 8; ++j)                     // C: VGPR j -> M=j(+8), N=lane%16
    h[((size_t)b * MPAD + j + roff) * DIN + n] = c[j] + bias;
}

// ----------------------------- batch-dim mean/var over the valid 20480 rows -
__global__ __launch_bounds__(128)
void k_reduce1(const float* __restrict__ h, float* __restrict__ gsum,
               float* __restrict__ gsq) {
  const int b = blockIdx.x, c = threadIdx.x * 4;
  float4 s = {0, 0, 0, 0}, q = {0, 0, 0, 0};
  for (int j = 0; j < SEL; ++j) {
    const float4 x = *(const float4*)(h + ((size_t)b * MPAD + j) * DIN + c);
    s.x += x.x; s.y += x.y; s.z += x.z; s.w += x.w;
    q.x += x.x * x.x; q.y += x.y * x.y; q.z += x.z * x.z; q.w += x.w * x.w;
  }
  atomicAdd(&gsum[c + 0], s.x); atomicAdd(&gsum[c + 1], s.y);
  atomicAdd(&gsum[c + 2], s.z); atomicAdd(&gsum[c + 3], s.w);
  atomicAdd(&gsq[c + 0],  q.x); atomicAdd(&gsq[c + 1],  q.y);
  atomicAdd(&gsq[c + 2],  q.z); atomicAdd(&gsq[c + 3],  q.w);
}

__global__ void k_reduce2(const float* __restrict__ gsum, const float* __restrict__ gsq,
                          const float* __restrict__ g0, const float* __restrict__ be0,
                          float* __restrict__ nscale, float* __restrict__ nbias) {
  const int c = blockIdx.x * 256 + threadIdx.x;
  if (c >= DIN) return;
  const float inv_n = 1.0f / (float)(NB * SEL);
  const float mean = gsum[c] * inv_n;
  const float var  = gsq[c] * inv_n - mean * mean;
  const float sc   = g0[c] * rsqrtf(var + 1e-5f);
  nscale[c] = sc;
  nbias[c]  = be0[c] - mean * sc;
}

// ---- GEMM2: out = relu(norm(h)) @ W1^T + b1 + f16(feats16 @ Wlin16^T + blin),
//      raw h tile staged via async global->LDS copies (overlapped with the
//      feats f16 conversion), fused masked max-pool over SELECT rows ---------
__global__ __launch_bounds__(256)
void k_gemm2(const float* __restrict__ h, const float* __restrict__ feats,
             const float* __restrict__ W1, const float* __restrict__ b1,
             const _Float16* __restrict__ Wlin16, const float* __restrict__ blin,
             const float* __restrict__ nscale, const float* __restrict__ nbias,
             const int* __restrict__ lv, float* __restrict__ out) {
  const int b = blockIdx.x;
  __shared__ float    lRaw[MPAD * DIN];           // raw h tile (async dest)
  __shared__ float    lH[MPAD * LDSF];            // normalized, padded stride
  __shared__ _Float16 lF[MPAD * LDSH];

  // 1) kick off async DMA of the raw h tile into LDS (b128 per lane)
  for (int i = threadIdx.x; i < MPAD * (DIN / 4); i += 256) {
    const int r = i >> 7, c = (i & 127) << 2;
    const unsigned lds = lds_off(&lRaw[r * DIN + c]);
    const unsigned long long g =
        (unsigned long long)(size_t)(h + ((size_t)b * MPAD + r) * DIN + c);
    asm volatile("global_load_async_to_lds_b128 %0, %1, off"
                 :: "v"(lds), "v"(g) : "memory");
  }
  // 2) overlap: feats -> f16 LDS tile while the async copies are in flight
  for (int i = threadIdx.x; i < MPAD * (DIN / 4); i += 256) {
    const int r = i >> 7, c = (i & 127) << 2;
    const float4 f = *(const float4*)(feats + ((size_t)b * MPAD + r) * DIN + c);
    _Float16* dst = &lF[r * LDSH + c];
    dst[0] = (_Float16)f.x; dst[1] = (_Float16)f.y;
    dst[2] = (_Float16)f.z; dst[3] = (_Float16)f.w;
  }
  asm volatile("s_wait_asynccnt 0x0" ::: "memory");
  __syncthreads();
  // 3) normalize + ReLU from LDS into the padded A tile
  for (int i = threadIdx.x; i < MPAD * (DIN / 4); i += 256) {
    const int r = i >> 7, c = (i & 127) << 2;
    const float4 x = *(const float4*)(&lRaw[r * DIN + c]);
    const float4 s = *(const float4*)(nscale + c);
    const float4 t = *(const float4*)(nbias + c);
    float4 y;
    y.x = fmaxf(x.x * s.x + t.x, 0.f); y.y = fmaxf(x.y * s.y + t.y, 0.f);
    y.z = fmaxf(x.z * s.z + t.z, 0.f); y.w = fmaxf(x.w * s.w + t.w, 0.f);
    *(float4*)(&lH[r * LDSF + c]) = y;
  }
  __syncthreads();

  const int wave = threadIdx.x >> 5, lane = threadIdx.x & 31;
  const int nt = blockIdx.y * 8 + wave;           // 64 n-tiles total
  const int n  = nt * 16 + (lane & 15);
  const int m  = lane & 15;
  const int k0 = (lane < 16) ? 0 : 2;
  const int hi = (lane < 16) ? 0 : 1;
  const float*    w1row = W1     + (size_t)n * DIN;
  const _Float16* wlrow = Wlin16 + (size_t)n * DIN;

  v8f c1 = {};   // f32 path: hn @ W1^T
  for (int kb = 0; kb < DIN; kb += 4) {
    if ((kb & 63) == 0) __builtin_prefetch(w1row + kb + 64, 0, 3);
#ifdef HAVE_WMMA_F32X4
    v2f a  = *(const v2f*)(&lH[m * LDSF + kb + k0]);
    v2f bb = *(const v2f*)(w1row + kb + k0);
    c1 = __builtin_amdgcn_wmma_f32_16x16x4_f32(false, a, false, bb, (short)0, c1,
                                               false, false);
#else
    const int roff = (lane < 16) ? 0 : 8;
    for (int kk = 0; kk < 4; ++kk) {
      const float bv = w1row[kb + kk];
      for (int j = 0; j < 8; ++j) c1[j] += lH[(j + roff) * LDSF + kb + kk] * bv;
    }
#endif
  }

  v8f c2 = {};   // f16 path: feats16 @ Wlin16^T (f32 accumulate)
  for (int kb = 0; kb < DIN; kb += 32) {
    __builtin_prefetch(wlrow + kb + 32, 0, 3);
    // A 16x32 f16 lane layout: lo lanes K[kb..kb+15]; hi lanes K[kb+16..kb+31]
    v8h a0 = *(const v8h*)(&lF[m * LDSH + kb + 16 * hi]);
    v8h a1 = *(const v8h*)(&lF[m * LDSH + kb + 16 * hi + 8]);
    v16h a = __builtin_shufflevector(a0, a1, 0, 1, 2, 3, 4, 5, 6, 7,
                                     8, 9, 10, 11, 12, 13, 14, 15);
    // B 32x16 f16: lane half selects K base kb(+16); 16 consecutive f16
    v16h bv = *(const v16h*)(wlrow + kb + 16 * hi);
    c2 = __builtin_amdgcn_wmma_f32_16x16x32_f16(false, a, false, bv, (short)0, c2,
                                                false, false);
  }

  // epilogue: bias adds (cap rounded through f16 like the reference cast),
  // masked max over the SELECT rows inside this C tile, cross-half shuffle
  const float    bias1 = b1[n];
  const _Float16 biasL = (_Float16)blin[n];
  const int lvb = lv[b];
  const int q   = lvb < 0 ? 0 : (lvb > SEL ? SEL : lvb);
  const int roff = (lane < 16) ? 0 : 8;
  float vmax = -INFINITY;
  for (int j = 0; j < 8; ++j) {
    const float cap = (float)(_Float16)((_Float16)c2[j] + biasL);
    const float val = c1[j] + bias1 + cap;
    if ((j + roff) < q) vmax = fmaxf(vmax, val);
  }
  const float other = __shfl_xor(vmax, 16, 32);   // combine M halves (wave32)
  vmax = fmaxf(vmax, other);
  if (lane < 16) out[(size_t)b * DEMB + n] = vmax;
}

// ---------------------------------------------------------------------------
extern "C" void kernel_launch(void* const* d_in, const int* in_sizes, int n_in,
                              void* d_out, int out_size, void* d_ws, size_t ws_size,
                              hipStream_t stream) {
  const float* emb  = (const float*)d_in[0];
  const int*   ids  = (const int*)  d_in[1];
  const float* attn = (const float*)d_in[2];
  const float* Wlin = (const float*)d_in[3];
  const float* blin = (const float*)d_in[4];
  const float* W0   = (const float*)d_in[5];
  const float* b0   = (const float*)d_in[6];
  const float* g0   = (const float*)d_in[7];
  const float* be0  = (const float*)d_in[8];
  const float* W1   = (const float*)d_in[9];
  const float* b1   = (const float*)d_in[10];
  float* out = (float*)d_out;

  char* ws = (char*)d_ws;
  const size_t featsOff = 0;                                  // 2048*16*512 f32 = 64 MB
  const size_t hOff     = featsOff + (size_t)NB * MPAD * DIN * 4;   // 64 MB
  const size_t wlinOff  = hOff     + (size_t)NB * MPAD * DIN * 4;   // 1 MB
  const size_t gsumOff  = wlinOff  + (size_t)DEMB * DIN * 2;
  const size_t gsqOff   = gsumOff  + DIN * 4;
  const size_t nscOff   = gsqOff   + DIN * 4;
  const size_t nbOff    = nscOff   + DIN * 4;
  const size_t lvOff    = nbOff    + DIN * 4;

  float*    feats  = (float*)(ws + featsOff);
  float*    hbuf   = (float*)(ws + hOff);
  _Float16* Wlin16 = (_Float16*)(ws + wlinOff);
  float*    gsum   = (float*)(ws + gsumOff);
  float*    gsq    = (float*)(ws + gsqOff);
  float*    nsc    = (float*)(ws + nscOff);
  float*    nb     = (float*)(ws + nbOff);
  int*      lv     = (int*)(ws + lvOff);

  k_prep<<<(DEMB * DIN + 255) / 256, 256, 0, stream>>>(Wlin, Wlin16, gsum, gsq);
  k_topk_gather<<<NB, 128, 0, stream>>>(emb, ids, attn, feats, lv);
  k_gemm1<<<dim3(NB, 4), 256, 0, stream>>>(feats, W0, b0, hbuf);
  k_reduce1<<<NB, 128, 0, stream>>>(hbuf, gsum, gsq);
  k_reduce2<<<2, 256, 0, stream>>>(gsum, gsq, g0, be0, nsc, nb);
  k_gemm2<<<dim3(NB, 8), 256, 0, stream>>>(hbuf, feats, W1, b1, Wlin16, blin,
                                           nsc, nb, lv, out);
}
